// Tiled_PIM_layer_64828236365983
// MI455X (gfx1250) — matrix operand order repaired and checked by reading, not compiled, not run on hardware
//
#include <hip/hip_runtime.h>

// CDNA5 / gfx1250: wave32, WMMA 16x16x32 bf16 -> f32 accumulate.
typedef __attribute__((ext_vector_type(16))) __bf16 v16bf;
typedef __attribute__((ext_vector_type(8)))  float  v8f;

constexpr int K_DIM = 2048;   // 8 tiles * 256 (r,i)
constexpr int N_DIM = 2048;   // 8 tiles * 256 (j,o)
constexpr int BM = 128, BN = 128, BK = 32;
constexpr int LDA = 40;       // padded bf16 row stride for A tiles (conflict-free 16B frag reads)
constexpr int LDB = 144;      // padded bf16 row stride for B tiles (min 4-way on 32B frag reads)

union Frag { v16bf bf; uint4 q[2]; };

__device__ __forceinline__ unsigned int pack_hi(float a, float b) {
  // dest[15:0] = a[31:16], dest[31:16] = b[31:16] -> single v_perm_b32
  // perm(src0, src1, sel): sel bytes 0-3 pick from src1, 4-7 from src0
  return __builtin_amdgcn_perm(__float_as_uint(b), __float_as_uint(a), 0x07060302u);
}
__device__ __forceinline__ float hi_part(float a) {
  return __uint_as_float(__float_as_uint(a) & 0xffff0000u);
}
__device__ __forceinline__ float lo_part(float a) { return a - hi_part(a); }

__global__ __launch_bounds__(256)
void pim_gemm_bf16x3(const float* __restrict__ X,
                     const float* __restrict__ W,
                     float* __restrict__ C) {
  // double-buffered tiles (~76 KB total)
  __shared__ __align__(16) unsigned short Ah[2][BM * LDA];  // A hi-bf16, [M][K]
  __shared__ __align__(16) unsigned short Al[2][BM * LDA];  // A lo-bf16
  __shared__ __align__(16) unsigned short Bh[2][BK * LDB];  // B hi-bf16, [K][N]
  __shared__ __align__(16) unsigned short Bl[2][BK * LDB];  // B lo-bf16

  const int t    = threadIdx.x;
  const int lane = t & 31;
  const int wid  = t >> 5;
  const int wm   = wid & 3;     // 4 waves along M, 32 rows each
  const int wn   = wid >> 2;    // 2 waves along N, 64 cols each

  const int bm = blockIdx.y * BM;
  const int bn = blockIdx.x * BN;
  const int j  = bn >> 8;       // which 256-wide W column tile
  const int o0 = bn & 255;

  const int srow  = t >> 3;         // 0..31
  const int scol  = (t & 7) << 2;   // 0..28 step 4 (K within tile)
  const int obase = srow << 2;      // 0..124 step 4 (N within tile, for B staging)

  // ---- staging helpers (register double buffer) ----
  auto loadA = [&](int kb, float4* ra) {
    #pragma unroll
    for (int it = 0; it < 4; ++it)
      ra[it] = *(const float4*)(X + (size_t)(bm + srow + it * 32) * K_DIM + kb + scol);
  };
  auto loadB = [&](int kb, float4* rb) {
    const int r = kb >> 8, i0 = kb & 255;
    // W[r][j][o][i] flat: ((r*8+j)*256+o)*256+i ; i (==K) contiguous
    const float* Wt = W + ((size_t)((r * 8 + j) * 256 + o0) * 256 + i0);
    #pragma unroll
    for (int c = 0; c < 4; ++c)
      rb[c] = *(const float4*)(Wt + (size_t)(obase + c) * 256 + scol);
  };
  auto storeA = [&](int p, const float4* ra) {
    #pragma unroll
    for (int it = 0; it < 4; ++it) {
      float4 v = ra[it];
      uint2 h, l;
      h.x = pack_hi(v.x, v.y);
      h.y = pack_hi(v.z, v.w);
      l.x = pack_hi(lo_part(v.x), lo_part(v.y));
      l.y = pack_hi(lo_part(v.z), lo_part(v.w));
      *(uint2*)&Ah[p][(srow + it * 32) * LDA + scol] = h;
      *(uint2*)&Al[p][(srow + it * 32) * LDA + scol] = l;
    }
  };
  auto storeB = [&](int p, const float4* rb) {
    // rb[c] holds W rows o=obase+c, cols i=scol..scol+3 -> 4x4 register transpose,
    // store 4 consecutive n as one b64 per k-row
    float v[4][4];
    #pragma unroll
    for (int c = 0; c < 4; ++c) {
      v[c][0] = rb[c].x; v[c][1] = rb[c].y; v[c][2] = rb[c].z; v[c][3] = rb[c].w;
    }
    #pragma unroll
    for (int d = 0; d < 4; ++d) {
      uint2 h, l;
      h.x = pack_hi(v[0][d], v[1][d]);
      h.y = pack_hi(v[2][d], v[3][d]);
      l.x = pack_hi(lo_part(v[0][d]), lo_part(v[1][d]));
      l.y = pack_hi(lo_part(v[2][d]), lo_part(v[3][d]));
      *(uint2*)&Bh[p][(scol + d) * LDB + obase] = h;
      *(uint2*)&Bl[p][(scol + d) * LDB + obase] = l;
    }
  };

  v8f acc[2][4] = {};
  float4 ra[4], rb[4];

  // prologue: stage first K-tile
  loadA(0, ra);
  loadB(0, rb);
  storeA(0, ra);
  storeB(0, rb);

  int p = 0;
  for (int kb = 0; kb < K_DIM; kb += BK) {
    __syncthreads();  // buffer p staged & previous compute done

    const bool nxt = (kb + BK) < K_DIM;
    if (nxt) {        // issue next tile's global loads; latency hides under WMMA below
      loadA(kb + BK, ra);
      loadB(kb + BK, rb);
    }

    // ---- A fragments (ISA 16-bit A 16x32: M=lane%16; lanes<16 K 0-7/16-23,
    //      lanes>=16 K 8-15/24-31; consecutive K packed into VGPR halves) ----
    Frag ah[2], al[2];
    #pragma unroll
    for (int mt = 0; mt < 2; ++mt) {
      const int mrow = wm * 32 + mt * 16 + (lane & 15);
      const int k0   = (lane & 16) ? 8 : 0;
      const unsigned short* pa = &Ah[p][mrow * LDA + k0];
      const unsigned short* pl = &Al[p][mrow * LDA + k0];
      ah[mt].q[0] = *(const uint4*)(pa);
      ah[mt].q[1] = *(const uint4*)(pa + 16);
      al[mt].q[0] = *(const uint4*)(pl);
      al[mt].q[1] = *(const uint4*)(pl + 16);
    }

    // ---- B fragments (N = lane%16; lanes<16 K 0-7/16-23, lanes>=16 K 8-15/24-31) ----
    Frag bh[4], bl[4];
    #pragma unroll
    for (int nt = 0; nt < 4; ++nt) {
      const int n0 = wn * 64 + nt * 16;
      const int kb0 = (lane & 16) ? 8 : 0;
      const unsigned short* pb = &Bh[p][kb0 * LDB + n0 + (lane & 15)];
      const unsigned short* pq = &Bl[p][kb0 * LDB + n0 + (lane & 15)];
      #pragma unroll
      for (int e = 0; e < 4; ++e) {
        (&bh[nt].q[0].x)[e] = pb[(2 * e) * LDB] | ((unsigned)pb[(2 * e + 1) * LDB] << 16);
        (&bh[nt].q[1].x)[e] = pb[(16 + 2 * e) * LDB] | ((unsigned)pb[(17 + 2 * e) * LDB] << 16);
        (&bl[nt].q[0].x)[e] = pq[(2 * e) * LDB] | ((unsigned)pq[(2 * e + 1) * LDB] << 16);
        (&bl[nt].q[1].x)[e] = pq[(16 + 2 * e) * LDB] | ((unsigned)pq[(17 + 2 * e) * LDB] << 16);
      }
    }

    // ---- bf16x3: hi*hi + hi*lo + lo*hi (lo*lo dropped, ~2^-16 rel err) ----
    #pragma unroll
    for (int mt = 0; mt < 2; ++mt) {
      #pragma unroll
      for (int nt = 0; nt < 4; ++nt) {
        acc[mt][nt] = __builtin_amdgcn_wmma_f32_16x16x32_bf16(
            false, ah[mt].bf, false, bh[nt].bf, (short)0, acc[mt][nt], false, false);
        acc[mt][nt] = __builtin_amdgcn_wmma_f32_16x16x32_bf16(
            false, ah[mt].bf, false, bl[nt].bf, (short)0, acc[mt][nt], false, false);
        acc[mt][nt] = __builtin_amdgcn_wmma_f32_16x16x32_bf16(
            false, al[mt].bf, false, bh[nt].bf, (short)0, acc[mt][nt], false, false);
      }
    }

    if (nxt) {        // convert+store next tile into the other buffer (waits loads here)
      storeA(p ^ 1, ra);
      storeB(p ^ 1, rb);
    }
    p ^= 1;
  }

  // ---- store D (ISA 16x16 f32 D layout: N = lane%16, M = g + 8*(lane/16)) ----
  const int mrow0 = bm + wm * 32 + ((lane & 16) ? 8 : 0);
  const int ncol0 = bn + wn * 64 + (lane & 15);
  #pragma unroll
  for (int mt = 0; mt < 2; ++mt) {
    #pragma unroll
    for (int nt = 0; nt < 4; ++nt) {
      #pragma unroll
      for (int g = 0; g < 8; ++g) {
        C[(size_t)(mrow0 + mt * 16 + g) * N_DIM + ncol0 + nt * 16] = acc[mt][nt][g];
      }
    }
  }
}

extern "C" void kernel_launch(void* const* d_in, const int* in_sizes, int n_in,
                              void* d_out, int out_size, void* d_ws, size_t ws_size,
                              hipStream_t stream) {
  (void)in_sizes; (void)n_in; (void)out_size; (void)d_ws; (void)ws_size;
  const float* X = (const float*)d_in[0];   // [8192, 2048] f32
  const float* W = (const float*)d_in[1];   // [8, 8, 256, 256] f32
  float* C = (float*)d_out;                 // [8192, 2048] f32
  dim3 grid(N_DIM / BN, 8192 / BM);         // (16, 64)
  pim_gemm_bf16x3<<<grid, 256, 0, stream>>>(X, W, C);
}
